// Rank_2989297238340
// MI455X (gfx1250) — matrix operand order repaired and checked by reading, not compiled, not run on hardware
//
#include <hip/hip_runtime.h>

#define NROWS 32768
#define DIM   512
#define HID   256

typedef __attribute__((ext_vector_type(16))) __bf16        v16bf;
typedef __attribute__((ext_vector_type(8)))  float         v8f;
typedef __attribute__((ext_vector_type(4)))  unsigned int  u32x4;
typedef __attribute__((ext_vector_type(8)))  unsigned int  u32x8;

union BFrag { v16bf v; unsigned int u[8]; };

// Truncating hi/lo bf16 split of an fp32 pair. hi + lo reproduces x to ~2^-17
// relative; the residual (x - hi) is exact in fp32.
__device__ __forceinline__ void split2(float f0, float f1,
                                       unsigned int& hi, unsigned int& lo) {
  unsigned int u0 = __float_as_uint(f0), u1 = __float_as_uint(f1);
  unsigned int h0 = u0 & 0xFFFF0000u,    h1 = u1 & 0xFFFF0000u;
  float r0 = f0 - __uint_as_float(h0);
  float r1 = f1 - __uint_as_float(h1);
  hi = (h0 >> 16) | h1;
  lo = (__float_as_uint(r0) >> 16) | (__float_as_uint(r1) & 0xFFFF0000u);
}

__device__ __forceinline__ float gelu_tanh(float x) {
  const float c0 = 0.7978845608028654f;   // sqrt(2/pi)
  const float c1 = 0.044715f;
  float t = tanhf(c0 * (x + c1 * x * x * x));
  return 0.5f * x * (1.0f + t);
}

// Issue a 1-D TDM copy of 4096 dwords (16 KB) global -> LDS.
__device__ __forceinline__ void tdm_load_slice(const unsigned int* gsrc,
                                               unsigned int ldsOff) {
  unsigned long long ga = (unsigned long long)(uintptr_t)gsrc;
  u32x4 g0;
  g0[0] = 1u;                                    // count=1, user D#
  g0[1] = ldsOff;                                // lds_addr
  g0[2] = (unsigned int)(ga & 0xFFFFFFFFu);      // global_addr[31:0]
  g0[3] = (unsigned int)((ga >> 32) & 0x01FFFFFFu) | (2u << 30); // type=2
  u32x8 g1;
  g1[0] = 2u << 16;                  // data_size=4B; mask=0; no pad/iter
  g1[1] = 0x1000u << 16;             // tensor_dim0[15:0] = 4096
  g1[2] = 1u << 16;                  // tensor_dim0 hi = 0; tensor_dim1 = 1
  g1[3] = 0x1000u << 16;             // tensor_dim1 hi = 0; tile_dim0 = 4096
  g1[4] = 1u;                        // tile_dim1 = 1, tile_dim2 = 0
  g1[5] = 4096u;                     // tensor_dim0_stride[31:0]
  g1[6] = 0u;
  g1[7] = 0u;
  u32x4 g2 = {0u, 0u, 0u, 0u};
  u32x4 g3 = {0u, 0u, 0u, 0u};
  asm volatile("tensor_load_to_lds %0, %1, %2, %3"
               :
               : "s"(g0), "s"(g1), "s"(g2), "s"(g3)
               : "memory");
}

// ---------------------------------------------------------------------------
// Kernel 0: one-time pack of W1 (fp32 [512][256]) into bf16 hi/lo B-fragment
// slices, contiguous 16 KB per (pass, kt):
//   w1frag[(pass*16+kt)*4096 + plane*2048 + ntl*256 + lane*8 + v]
// B-frag layout (wave32, 16-bit B 32x16): lanes 0-15 hold K=0..7,16..23 (2/VGPR),
// lanes 16-31 hold K=8..15,24..31; N = lane%16; pass selects n-tiles 0-7 / 8-15.
// ---------------------------------------------------------------------------
__global__ __launch_bounds__(256) void w1_pack_kernel(
    const float* __restrict__ W1, unsigned int* __restrict__ w1frag) {
  int t    = blockIdx.x * 256 + threadIdx.x;   // [0, 65536)
  int v    = t & 7;
  int lane = (t >> 3) & 31;
  int ntl  = (t >> 8) & 7;
  int kt   = (t >> 11) & 15;
  int pass = t >> 15;
  int n    = (pass * 8 + ntl) * 16 + (lane & 15);
  int k0   = kt * 32 + ((lane < 16) ? 0 : 8) + ((v < 4) ? 2 * v : 16 + 2 * (v - 4));
  float w0 = W1[(size_t)k0 * HID + n];
  float w1 = W1[(size_t)(k0 + 1) * HID + n];
  unsigned int hi, lo;
  split2(w0, w1, hi, lo);
  size_t base = (size_t)(pass * 16 + kt) * 4096 + ntl * 256 + lane * 8 + v;
  w1frag[base]        = hi;
  w1frag[base + 2048] = lo;
}

// ---------------------------------------------------------------------------
// Kernel 1: a[i] = gelu(input @ W1 + b1) @ W2 + b2 via bf16 hi/lo split WMMA.
// Block = 256 threads = 8 waves; wave owns 16 rows; block owns 128 rows.
// Two passes over HID halves (acc = 64 VGPRs); per K-step wave 0 streams the
// next 16 KB W1-fragment slice into the other LDS buffer with the TDM while
// all 8 waves run 8 n-tiles x 3 WMMAs on the current buffer.
// ---------------------------------------------------------------------------
__global__ __launch_bounds__(256) void mlp_score_kernel(
    const float* __restrict__ inp, const unsigned int* __restrict__ w1frag,
    const float* __restrict__ b1,  const float* __restrict__ W2,
    const float* __restrict__ b2,  float* __restrict__ aOut) {
  __shared__ unsigned int sB[2][2][8][32][8];  // [buf][plane][ntl][lane][v]=32KB

  const int tid  = threadIdx.x;
  const int lane = tid & 31;
  const int wave = tid >> 5;
  const int m    = lane & 15;
  const int rowBase = blockIdx.x * 128 + wave * 16;
  const int row  = rowBase + m;
  const int kb   = (lane < 16) ? 0 : 8;        // A-frag K base per lane half

  const unsigned int ldsOff = (unsigned int)(uintptr_t)(void*)&sB[0][0][0][0][0];

  float part[8];
#pragma unroll
  for (int r = 0; r < 8; ++r) part[r] = 0.0f;

#pragma unroll 1
  for (int p = 0; p < 2; ++p) {
    v8f acc[8];
#pragma unroll
    for (int i = 0; i < 8; ++i) acc[i] = v8f{};

    if (wave == 0)   // prologue: slice (p, kt=0) -> buffer 0
      tdm_load_slice(w1frag + (size_t)(p * 16) * 4096, ldsOff);

#pragma unroll 1
    for (int kt = 0; kt < 16; ++kt) {
      const int buf = kt & 1;
      if (wave == 0) __builtin_amdgcn_s_wait_tensorcnt(0);
      __syncthreads();                         // slice (p,kt) visible to all
      if (wave == 0 && kt + 1 < 16)            // overlap next DMA with compute
        tdm_load_slice(w1frag + (size_t)(p * 16 + kt + 1) * 4096,
                       ldsOff + (unsigned int)((buf ^ 1) * 16384));

      // ---- A fragment: this wave's 16 rows, K slice [kt*32, kt*32+32) ----
      const float* ar = inp + (size_t)row * DIM + kt * 32 + kb;
      float4 x0 = *(const float4*)(ar + 0);    // K = kb+0..3
      float4 x1 = *(const float4*)(ar + 4);    // K = kb+4..7
      float4 x2 = *(const float4*)(ar + 16);   // K = kb+16..19
      float4 x3 = *(const float4*)(ar + 20);   // K = kb+20..23

      BFrag ah, al;
      split2(x0.x, x0.y, ah.u[0], al.u[0]);
      split2(x0.z, x0.w, ah.u[1], al.u[1]);
      split2(x1.x, x1.y, ah.u[2], al.u[2]);
      split2(x1.z, x1.w, ah.u[3], al.u[3]);
      split2(x2.x, x2.y, ah.u[4], al.u[4]);
      split2(x2.z, x2.w, ah.u[5], al.u[5]);
      split2(x3.x, x3.y, ah.u[6], al.u[6]);
      split2(x3.z, x3.w, ah.u[7], al.u[7]);

      // ---- 8 n-tiles, 3 WMMAs each (hi*hi + hi*lo + lo*hi) ----
#pragma unroll
      for (int ntl = 0; ntl < 8; ++ntl) {
        BFrag bh, bl;
        const unsigned int* ph = &sB[buf][0][ntl][lane][0];
        const unsigned int* pl = &sB[buf][1][ntl][lane][0];
#pragma unroll
        for (int v = 0; v < 8; ++v) { bh.u[v] = ph[v]; bl.u[v] = pl[v]; }
        acc[ntl] = __builtin_amdgcn_wmma_f32_16x16x32_bf16(
            false, ah.v, false, bh.v, (short)0, acc[ntl], false, false);
        acc[ntl] = __builtin_amdgcn_wmma_f32_16x16x32_bf16(
            false, ah.v, false, bl.v, (short)0, acc[ntl], false, false);
        acc[ntl] = __builtin_amdgcn_wmma_f32_16x16x32_bf16(
            false, al.v, false, bh.v, (short)0, acc[ntl], false, false);
      }
      // next iteration's wait+barrier fences buffer reuse; no 2nd barrier
    }
    __syncthreads();   // don't let wave 0 of next pass overwrite early

    // ---- epilogue half: +b1, GELU, dot with W2 into part[] ----
#pragma unroll
    for (int ntl = 0; ntl < 8; ++ntl) {
      int n = (p * 8 + ntl) * 16 + m;          // C/D layout: N = lane%16
      float bn = b1[n];
      float w2 = W2[n];
#pragma unroll
      for (int r = 0; r < 8; ++r) {            // VGPR r: rows r / r+8
        float h = gelu_tanh(acc[ntl][r] + bn);
        part[r] += h * w2;
      }
    }
  }

  // ---- 16-lane butterfly reduction over N, then +b2 and store ----
#pragma unroll
  for (int off = 1; off < 16; off <<= 1) {
#pragma unroll
    for (int r = 0; r < 8; ++r) part[r] += __shfl_xor(part[r], off, 32);
  }
  if (m == 0) {
    float bias = b2[0];
    int mb = rowBase + ((lane < 16) ? 0 : 8);
#pragma unroll
    for (int r = 0; r < 8; ++r) aOut[mb + r] = part[r] + bias;
  }
}

// ---------------------------------------------------------------------------
// Kernel 2: z[i][j] = ((w_j + a_i) - w_j) * input[i][j] / a_i,  w_j = 512 - j.
// Replicates the reference's fp32 rounding of the degenerate soft-sort.
// ---------------------------------------------------------------------------
__global__ __launch_bounds__(128) void softsort_st_kernel(
    const float* __restrict__ inp, const float* __restrict__ a,
    float* __restrict__ out) {
  const int row = blockIdx.x;
  const int t   = threadIdx.x;
  const float av = a[row];
  const size_t base = (size_t)row * DIM + (size_t)t * 4;
  float4 x = *(const float4*)(inp + base);
  float w0 = (float)(DIM - t * 4);
  float w1 = w0 - 1.0f, w2 = w0 - 2.0f, w3 = w0 - 3.0f;
  float y0 = (w0 + av) - w0;
  float y1 = (w1 + av) - w1;
  float y2 = (w2 + av) - w2;
  float y3 = (w3 + av) - w3;
  float4 z;
  z.x = (y0 * x.x) / av;
  z.y = (y1 * x.y) / av;
  z.z = (y2 * x.z) / av;
  z.w = (y3 * x.w) / av;
  *(float4*)(out + base) = z;
}

extern "C" void kernel_launch(void* const* d_in, const int* in_sizes, int n_in,
                              void* d_out, int out_size, void* d_ws, size_t ws_size,
                              hipStream_t stream) {
  (void)in_sizes; (void)n_in; (void)out_size; (void)ws_size;
  const float* inp = (const float*)d_in[0];
  const float* W1  = (const float*)d_in[1];
  const float* b1  = (const float*)d_in[2];
  const float* W2  = (const float*)d_in[3];
  const float* b2  = (const float*)d_in[4];

  float*        aScratch = (float*)d_ws;                          // 128 KB
  unsigned int* w1frag   = (unsigned int*)((char*)d_ws + 131072); // 512 KB
  float*        out      = (float*)d_out;

  w1_pack_kernel<<<256, 256, 0, stream>>>(W1, w1frag);
  mlp_score_kernel<<<NROWS / 128, 256, 0, stream>>>(inp, w1frag, b1, W2, b2,
                                                    aScratch);
  softsort_st_kernel<<<NROWS, 128, 0, stream>>>(inp, aScratch, out);
}
